// Critic_26190710571292
// MI455X (gfx1250) — matrix operand order repaired and checked by reading, not compiled, not run on hardware
//
#include <hip/hip_runtime.h>

typedef __bf16 bf16;
typedef unsigned int u32;
typedef __attribute__((ext_vector_type(4)))  u32   u32x4;
typedef __attribute__((ext_vector_type(16))) bf16  v16bf;
typedef __attribute__((ext_vector_type(8)))  float v8f;
typedef __attribute__((ext_vector_type(4)))  float f32x4;

#define T_DIM 128
#define B_DIM 512
#define D_DIM 512
#define M_MEM 6
#define J_DIM 134
#define H_DIM 2
#define DH_DIM 32

static __device__ __forceinline__ void wait_asynccnt0() {
#if defined(__has_builtin)
#if __has_builtin(__builtin_amdgcn_s_wait_asynccnt)
  __builtin_amdgcn_s_wait_asynccnt(0);
  return;
#else
  asm volatile("s_wait_asynccnt 0" ::: "memory");
  return;
#endif
#else
  asm volatile("s_wait_asynccnt 0" ::: "memory");
#endif
}

// issue a 16B global -> LDS async copy (per-lane), tracked on ASYNCcnt
static __device__ __forceinline__ void async_copy_b128(unsigned lds_off,
                                                       const void* gaddr) {
  asm volatile("global_load_async_to_lds_b128 %0, %1, off"
               :: "v"(lds_off), "v"(gaddr) : "memory");
}

// ---------------------------------------------------------------------------
// Generic bf16 WMMA GEMM:  C[M,N] = act( A0@B0t (+ A1@B1t) (+ C if accum) + b0 + b1 )
// A: [M,K] bf16 row-major.  B*t: [N,K] bf16 (pre-transposed weights).
// Optional bf16 mirror output Cbf with leading dim ldcb.
// Block tile 128 x (NT*32); 8 waves (4 along M x 2 along N), each wave
// 32 x (NT*16) => 2 x NT WMMA tiles. Double-buffered LDS; tiles staged with
// GLOBAL_LOAD_ASYNC_TO_LDS_B128 (ASYNCcnt) overlapping the WMMA block.
// The two (A,B) pairs are flattened into one virtual K-range.
// ---------------------------------------------------------------------------
#define GTM 128
#define GTK 32
#define LDS_STRIDE 40   // 32 halfs + 8 pad -> 80B row stride, conflict-free b128

union FragU { u32x4 u[2]; v16bf v; };

template<int NT, bool GUARD>
__global__ __launch_bounds__(256) void gemm_bf16_kernel(
    const bf16* __restrict__ A0, const bf16* __restrict__ B0, int K0,
    const bf16* __restrict__ A1, const bf16* __restrict__ B1, int K1,
    const float* __restrict__ bias0, const float* __restrict__ bias1,
    float* __restrict__ C, bf16* __restrict__ Cbf, int ldcb,
    int M, int N, int act, int accum)
{
  constexpr int BN  = NT * 32;          // 64 or 128
  constexpr int BCH = (BN * 4) / 256;   // B-tile 16B chunks per thread (1 or 2)
  __shared__ alignas(16) bf16 sA[2][GTM * LDS_STRIDE];
  __shared__ alignas(16) bf16 sB[2][BN  * LDS_STRIDE];

  const int tid  = threadIdx.x;
  const int lane = tid & 31;
  const int wave = tid >> 5;
  const int bm = blockIdx.y * GTM;
  const int bn = blockIdx.x * BN;
  const int wm = (wave & 3) * 32;
  const int wn = (wave >> 2) * (BN / 2);
  const int lr = lane & 15;
  const int hi = lane >> 4;

  const int nt0 = K0 / GTK;
  const int nt1 = (A1 != nullptr) ? (K1 / GTK) : 0;
  const int nt  = nt0 + nt1;

  const v8f vzero = {0.f,0.f,0.f,0.f,0.f,0.f,0.f,0.f};
  v8f acc[2][NT];
  #pragma unroll
  for (int i = 0; i < 2; ++i)
    #pragma unroll
    for (int j = 0; j < NT; ++j) acc[i][j] = vzero;

  // stage tile s into LDS buffer `buf` with async global->LDS copies
  auto stage_async = [&](int s, int buf) {
    const bf16* Ap; const bf16* Bp; int K; int kk;
    if (s < nt0) { Ap = A0; Bp = B0; K = K0; kk = s * GTK; }
    else         { Ap = A1; Bp = B1; K = K1; kk = (s - nt0) * GTK; }
    #pragma unroll
    for (int i = 0; i < 2; ++i) {
      int c = tid + i * 256;
      int r = c >> 2, cc = (c & 3) * 8;
      int gm = bm + r;
      bf16* lp = &sA[buf][r * LDS_STRIDE + cc];
      if (!GUARD || gm < M) {
        async_copy_b128((unsigned)(size_t)lp, Ap + (size_t)gm * K + kk + cc);
      } else {
        u32x4 z = {0u,0u,0u,0u};
        *(u32x4*)lp = z;
      }
    }
    #pragma unroll
    for (int i = 0; i < BCH; ++i) {
      int c = tid + i * 256;
      int r = c >> 2, cc = (c & 3) * 8;
      int gn = bn + r;
      bf16* lp = &sB[buf][r * LDS_STRIDE + cc];
      if (!GUARD || gn < N) {
        async_copy_b128((unsigned)(size_t)lp, Bp + (size_t)gn * K + kk + cc);
      } else {
        u32x4 z = {0u,0u,0u,0u};
        *(u32x4*)lp = z;
      }
    }
    // prefetch A one tile further ahead (global_prefetch_b8)
    int s2 = s + 1;
    if (s2 < nt) {
      const bf16* Ap2; int K2, kk2;
      if (s2 < nt0) { Ap2 = A0; K2 = K0; kk2 = s2 * GTK; }
      else          { Ap2 = A1; K2 = K1; kk2 = (s2 - nt0) * GTK; }
      int r = tid >> 2, cc = (tid & 3) * 8;
      int gm = bm + r;
      if (!GUARD || gm < M)
        __builtin_prefetch(Ap2 + (size_t)gm * K2 + kk2 + cc, 0, 1);
    }
  };

  stage_async(0, 0);
  wait_asynccnt0();

  for (int s = 0; s < nt; ++s) {
    const int buf = s & 1;
    __syncthreads();                  // LDS[buf] writes visible to all waves
    if (s + 1 < nt) stage_async(s + 1, 1 - buf);  // async copies in flight

    FragU fa[2]; FragU fb[NT];
    const bf16* baseA = sA[buf];
    const bf16* baseB = sB[buf];
    #pragma unroll
    for (int i = 0; i < 2; ++i) {     // A: K chunks at hi*8 and hi*8+16 (ISA 7.12.2)
      const bf16* p = &baseA[(wm + i*16 + lr) * LDS_STRIDE + hi * 8];
      fa[i].u[0] = *(const u32x4*)p;
      fa[i].u[1] = *(const u32x4*)(p + 16);
    }
    #pragma unroll
    for (int j = 0; j < NT; ++j) {    // B: K chunks at hi*8 and hi*8+16
      const bf16* p = &baseB[(wn + j*16 + lr) * LDS_STRIDE + hi * 8];
      fb[j].u[0] = *(const u32x4*)p;
      fb[j].u[1] = *(const u32x4*)(p + 16);
    }
    #pragma unroll
    for (int i = 0; i < 2; ++i)
      #pragma unroll
      for (int j = 0; j < NT; ++j)
        acc[i][j] = __builtin_amdgcn_wmma_f32_16x16x32_bf16(
            false, fa[i].v, false, fb[j].v, (short)0, acc[i][j], false, false);

    if (s + 1 < nt) wait_asynccnt0(); // my async writes landed before next barrier
  }

  // epilogue: C VGPR r -> row (r + hi*8), col = lane&15
  #pragma unroll
  for (int i = 0; i < 2; ++i) {
    #pragma unroll
    for (int j = 0; j < NT; ++j) {
      int col = bn + wn + j*16 + lr;
      if (GUARD && col >= N) continue;
      float badd = 0.f;
      if (bias0) badd += bias0[col];
      if (bias1) badd += bias1[col];
      #pragma unroll
      for (int r = 0; r < 8; ++r) {
        int row = bm + wm + i*16 + hi*8 + r;
        if (GUARD && row >= M) continue;
        float v = acc[i][j][r] + badd;
        if (accum) v += C[(size_t)row * N + col];
        if (act == 1) v = fmaxf(v, 0.f);
        if (C)   C[(size_t)row * N + col]      = v;
        if (Cbf) Cbf[(size_t)row * ldcb + col] = (bf16)v;
      }
    }
  }
}

// ---------------------------------------------------------------------------
// Weight transpose + f32->bf16:  Wt[n*K + k] = W[k*N + n]
// ---------------------------------------------------------------------------
__global__ void transpose_bf16_kernel(const float* __restrict__ W,
                                      bf16* __restrict__ Wt, int K, int N)
{
  int idx = blockIdx.x * 256 + threadIdx.x;
  if (idx >= K * N) return;
  int n = idx / K;
  int k = idx - n * K;
  Wt[idx] = (bf16)W[(size_t)k * N + n];
}

__global__ void cvt_bf16_kernel(const float* __restrict__ in,
                                bf16* __restrict__ out, long long n)
{
  long long i = (long long)blockIdx.x * 256 + threadIdx.x;
  if (i < n) out[i] = (bf16)in[i];
}

// ---------------------------------------------------------------------------
// LayerNorm over D=512, one row per block (128 threads x float4), bf16 out
// ---------------------------------------------------------------------------
__global__ __launch_bounds__(128) void ln_kernel(
    const float* __restrict__ X, const float* __restrict__ g,
    const float* __restrict__ b, bf16* __restrict__ out)
{
  __shared__ float red[4];
  size_t row = blockIdx.x;
  int t = threadIdx.x;
  f32x4 v = ((const f32x4*)(X + row * 512))[t];
  float s = v.x + v.y + v.z + v.w;
  for (int o = 16; o > 0; o >>= 1) s += __shfl_down(s, o, 32);
  if ((t & 31) == 0) red[t >> 5] = s;
  __syncthreads();
  float mean = (red[0] + red[1] + red[2] + red[3]) * (1.f / 512.f);
  float dx = v.x - mean, dy = v.y - mean, dz = v.z - mean, dw = v.w - mean;
  float q = dx*dx + dy*dy + dz*dz + dw*dw;
  for (int o = 16; o > 0; o >>= 1) q += __shfl_down(q, o, 32);
  __syncthreads();
  if ((t & 31) == 0) red[t >> 5] = q;
  __syncthreads();
  float var  = (red[0] + red[1] + red[2] + red[3]) * (1.f / 512.f);
  float rstd = rsqrtf(var + 1e-5f);
  f32x4 gg = ((const f32x4*)g)[t];
  f32x4 bb = ((const f32x4*)b)[t];
  bf16* o = out + row * 512 + t * 4;
  o[0] = (bf16)(dx * rstd * gg.x + bb.x);
  o[1] = (bf16)(dy * rstd * gg.y + bb.y);
  o[2] = (bf16)(dz * rstd * gg.z + bb.z);
  o[3] = (bf16)(dw * rstd * gg.w + bb.w);
}

// ---------------------------------------------------------------------------
// Relative position embedding projected: P[j, h*32+d] = pos_emb[j] @ W_p
// ---------------------------------------------------------------------------
__global__ void pos_kernel(const float* __restrict__ Wp, float* __restrict__ P)
{
  int j = blockIdx.x;
  int e = threadIdx.x;
  float pos = (float)(J_DIM - 1 - j);
  float acc = 0.f;
  for (int i = 0; i < 256; ++i) {
    float invf = expf(-0.0359778921f * (float)i);   // ln(10000)/256
    float a = pos * invf;
    acc = fmaf(sinf(a), Wp[i * 64 + e], acc);
    acc = fmaf(cosf(a), Wp[(i + 256) * 64 + e], acc);
  }
  P[j * 64 + e] = acc;
}

// ---------------------------------------------------------------------------
// TransformerXL attention per (b,h): online softmax over j<=i+M,
// rel-shifted position term uses p[j - i + T - 1].
// ---------------------------------------------------------------------------
#define KSTR 33
__global__ __launch_bounds__(128) void attn_kernel(
    const float* __restrict__ Q, const float* __restrict__ KV,
    const float* __restrict__ P, const float* __restrict__ U,
    const float* __restrict__ V, float* __restrict__ AV,
    bf16* __restrict__ AVbf, int B)
{
  __shared__ float kl[J_DIM * KSTR];
  __shared__ float vl[J_DIM * KSTR];
  __shared__ float pl[J_DIM * KSTR];
  int bh = blockIdx.x;
  int b = bh >> 1, h = bh & 1;
  int tid = threadIdx.x;
  for (int idx = tid; idx < J_DIM * DH_DIM; idx += 128) {
    int j = idx / DH_DIM, d = idx - j * DH_DIM;
    const float* kv = KV + ((size_t)j * B + b) * 128;
    kl[j * KSTR + d] = kv[h * DH_DIM + d];
    vl[j * KSTR + d] = kv[64 + h * DH_DIM + d];
    pl[j * KSTR + d] = P[j * 64 + h * DH_DIM + d];
  }
  __syncthreads();

  int i = tid;
  float qu[DH_DIM], qv[DH_DIM], accv[DH_DIM];
  const float* q = Q + ((size_t)i * B + b) * 64 + h * DH_DIM;
  #pragma unroll
  for (int d = 0; d < DH_DIM; ++d) {
    float qd = q[d];
    qu[d] = qd + U[h * DH_DIM + d];
    qv[d] = qd + V[h * DH_DIM + d];
    accv[d] = 0.f;
  }
  float m = -1e30f, l = 0.f;
  const float scale = 0.17677669529663687f;       // 1/sqrt(32)
  int jmax = i + M_MEM;                            // inclusive causal bound
  for (int j = 0; j <= jmax; ++j) {
    const float* kr = &kl[j * KSTR];
    const float* pr = &pl[(j - i + T_DIM - 1) * KSTR];
    float s = 0.f, sp = 0.f;
    #pragma unroll
    for (int d = 0; d < DH_DIM; ++d) { s = fmaf(qu[d], kr[d], s); sp = fmaf(qv[d], pr[d], sp); }
    s = (s + sp) * scale;
    float mn   = fmaxf(m, s);
    float corr = expf(m - mn);
    float pj   = expf(s - mn);
    l = l * corr + pj;
    const float* vr = &vl[j * KSTR];
    #pragma unroll
    for (int d = 0; d < DH_DIM; ++d) accv[d] = accv[d] * corr + pj * vr[d];
    m = mn;
  }
  float inv = 1.f / l;
  float* op = AV   + ((size_t)i * B + b) * 64 + h * DH_DIM;
  bf16*  ob = AVbf + ((size_t)i * B + b) * 64 + h * DH_DIM;
  #pragma unroll
  for (int d = 0; d < DH_DIM; ++d) { float o = accv[d] * inv; op[d] = o; ob[d] = (bf16)o; }
}

// ---------------------------------------------------------------------------
// Gate elementwise passes
// ---------------------------------------------------------------------------
__global__ void gelem1_kernel(const float* __restrict__ tr, float* __restrict__ tz,
                              const float* __restrict__ x, bf16* __restrict__ rxbf,
                              long long n)
{
  long long i = (long long)blockIdx.x * 256 + threadIdx.x;
  if (i >= n) return;
  float r = 1.f / (1.f + expf(-tr[i]));
  float z = 1.f / (1.f + expf(-(tz[i] - 0.1f)));    // BG = 0.1
  tz[i] = z;
  rxbf[i] = (bf16)(r * x[i]);
}

__global__ void gelem2_kernel(const float* __restrict__ z, const float* __restrict__ x,
                              const float* __restrict__ th, float* __restrict__ out,
                              long long n)
{
  long long i = (long long)blockIdx.x * 256 + threadIdx.x;
  if (i >= n) return;
  float h  = tanhf(th[i]);
  float zz = z[i];
  out[i] = (1.f - zz) * x[i] + zz * h;
}

// ---------------------------------------------------------------------------
// Mean over T (axis 0) of [T,B,D], write f32 + bf16
// ---------------------------------------------------------------------------
__global__ void pool_kernel(const float* __restrict__ X, float* __restrict__ ts,
                            bf16* __restrict__ tsbf, int T, size_t BD)
{
  size_t idx = (size_t)blockIdx.x * 256 + threadIdx.x;
  if (idx >= BD) return;
  float s = 0.f;
  for (int t = 0; t < T; ++t) s += X[(size_t)t * BD + idx];
  s *= (1.f / (float)T);
  ts[idx] = s;
  tsbf[idx] = (bf16)s;
}

__global__ void actcat_kernel(const float* __restrict__ act, bf16* __restrict__ cat)
{
  int idx = blockIdx.x * 256 + threadIdx.x;     // 512 * 64
  if (idx >= 512 * 64) return;
  int b = idx >> 6, a = idx & 63;
  cat[(size_t)b * 1088 + 1024 + a] = (bf16)act[idx];
}

__global__ void head4_kernel(const float* __restrict__ H3, const float* __restrict__ W,
                             const float* __restrict__ bias, float* __restrict__ out,
                             int Bn, int K)
{
  int b = blockIdx.x * 256 + threadIdx.x;
  if (b >= Bn) return;
  float acc = bias[0];
  for (int k = 0; k < K; ++k) acc = fmaf(H3[(size_t)b * K + k], W[k], acc);
  out[b] = fmaxf(acc, 0.f);
}

// ---------------------------------------------------------------------------
extern "C" void kernel_launch(void* const* d_in, const int* in_sizes, int n_in,
                              void* d_out, int out_size, void* d_ws, size_t ws_size,
                              hipStream_t stream)
{
  (void)in_sizes; (void)n_in; (void)out_size; (void)ws_size;
  const float* aug    = (const float*)d_in[0];
  const float* action = (const float*)d_in[1];
  const float* memin  = (const float*)d_in[2];
  const float* W_q    = (const float*)d_in[3];
  const float* W_kv   = (const float*)d_in[4];
  const float* W_p    = (const float*)d_in[5];
  const float* W_out  = (const float*)d_in[6];
  const float* u_     = (const float*)d_in[7];
  const float* v_     = (const float*)d_in[8];
  const float* ln1_g  = (const float*)d_in[9];
  const float* ln1_b  = (const float*)d_in[10];
  const float* ln2_g  = (const float*)d_in[11];
  const float* ln2_b  = (const float*)d_in[12];
  const float* ff_W1  = (const float*)d_in[13];
  const float* ff_b1  = (const float*)d_in[14];
  const float* ff_W2  = (const float*)d_in[15];
  const float* ff_b2  = (const float*)d_in[16];
  const float* g1_W   = (const float*)d_in[17];
  const float* g1_b   = (const float*)d_in[18];
  const float* g2_W   = (const float*)d_in[19];
  const float* g2_b   = (const float*)d_in[20];
  const float* d1_W   = (const float*)d_in[21];
  const float* d1_b   = (const float*)d_in[22];
  const float* d2_W   = (const float*)d_in[23];
  const float* d2_b   = (const float*)d_in[24];
  const float* d3_W   = (const float*)d_in[25];
  const float* d3_b   = (const float*)d_in[26];
  const float* d4_W   = (const float*)d_in[27];
  const float* d4_b   = (const float*)d_in[28];

  const size_t NBD   = (size_t)T_DIM * B_DIM * D_DIM;
  const size_t JBD   = (size_t)J_DIM * B_DIM * D_DIM;
  const size_t TB64  = (size_t)T_DIM * B_DIM * 64;
  const size_t JB128 = (size_t)J_DIM * B_DIM * 128;
  const size_t DD    = (size_t)D_DIM * D_DIM;

  char* base = (char*)d_ws;
  size_t off = 0;
  auto alloc = [&](size_t bytes) -> void* {
    off = (off + 255) & ~(size_t)255;
    void* p = base + off;
    off += bytes;
    return p;
  };

  bf16*  wq_t  = (bf16*)alloc((size_t)512*64*2);
  bf16*  wkv_t = (bf16*)alloc((size_t)512*128*2);
  bf16*  wout_t= (bf16*)alloc((size_t)64*512*2);
  bf16*  ff1_t = (bf16*)alloc((size_t)512*64*2);
  bf16*  ff2_t = (bf16*)alloc((size_t)64*512*2);
  bf16*  g1_t  = (bf16*)alloc((size_t)6*DD*2);
  bf16*  g2_t  = (bf16*)alloc((size_t)6*DD*2);
  bf16*  d1_t  = (bf16*)alloc((size_t)512*1024*2);
  bf16*  d2_t  = (bf16*)alloc((size_t)1088*512*2);
  bf16*  d3_t  = (bf16*)alloc((size_t)512*300*2);
  float* Pbuf  = (float*)alloc((size_t)J_DIM*64*4);
  bf16*  bigH0 = (bf16*)alloc(JBD*2);      // cat_bf, later xbf/rxbf
  bf16*  bigH1 = (bf16*)alloc(NBD*2);      // ybf / ln2bf / ybf2
  float* Qf    = (float*)alloc(TB64*4);
  float* KVf   = (float*)alloc(JB128*4);
  float* AVf   = (float*)alloc(TB64*4);
  bf16*  AVbf  = (bf16*)alloc(TB64*2);
  float* F0    = (float*)alloc(NBD*4);     // branch output y
  float* F1    = (float*)alloc(NBD*4);     // r-pre, later final gate2 out
  float* F2    = (float*)alloc(NBD*4);     // z
  float* F3    = (float*)alloc(NBD*4);     // h-pre
  float* F4    = (float*)alloc(NBD*4);     // gate1 output (residual)
  bf16*  ff1bf = (bf16*)alloc(TB64*2);
  float* ts    = (float*)alloc((size_t)512*512*4);
  bf16*  tsbf  = (bf16*)alloc((size_t)512*512*2);
  bf16*  cath  = (bf16*)alloc((size_t)512*1088*2);
  bf16*  h2bf  = (bf16*)alloc((size_t)512*512*2);
  float* h3    = (float*)alloc((size_t)512*300*4);

  auto tw = [&](const float* W, bf16* Wt, int K, int N) {
    int n = K * N;
    transpose_bf16_kernel<<<(n + 255) / 256, 256, 0, stream>>>(W, Wt, K, N);
  };
  tw(W_q,   wq_t,  512, 64);
  tw(W_kv,  wkv_t, 512, 128);
  tw(W_out, wout_t, 64, 512);
  tw(ff_W1, ff1_t, 512, 64);
  tw(ff_W2, ff2_t,  64, 512);
  for (int i = 0; i < 6; ++i) tw(g1_W + i * DD, g1_t + i * DD, 512, 512);
  for (int i = 0; i < 6; ++i) tw(g2_W + i * DD, g2_t + i * DD, 512, 512);
  tw(d1_W, d1_t, 512, 1024);
  tw(d2_W, d2_t, 1088, 512);
  tw(d3_W, d3_t, 512, 300);

  pos_kernel<<<J_DIM, 64, 0, stream>>>(W_p, Pbuf);

  long long MBD = (long long)M_MEM * B_DIM * D_DIM;
  cvt_bf16_kernel<<<(int)((MBD + 255) / 256), 256, 0, stream>>>(memin, bigH0, MBD);
  ln_kernel<<<T_DIM * B_DIM, 128, 0, stream>>>(aug, ln1_g, ln1_b, bigH0 + (size_t)MBD);

  auto gemm = [&](const bf16* A0, const bf16* B0, int K0,
                  const bf16* A1, const bf16* B1, int K1,
                  const float* b0, const float* b1,
                  float* Cc, bf16* Cb, int ldcb, int Mr, int Nc, int act, int accum) {
    if ((Nc & 127) == 0) {
      dim3 grd(Nc / 128, (Mr + GTM - 1) / GTM);
      gemm_bf16_kernel<4,false><<<grd, 256, 0, stream>>>(A0, B0, K0, A1, B1, K1,
          b0, b1, Cc, Cb, ldcb, Mr, Nc, act, accum);
    } else if ((Nc & 63) == 0) {
      dim3 grd(Nc / 64, (Mr + GTM - 1) / GTM);
      gemm_bf16_kernel<2,false><<<grd, 256, 0, stream>>>(A0, B0, K0, A1, B1, K1,
          b0, b1, Cc, Cb, ldcb, Mr, Nc, act, accum);
    } else {
      dim3 grd((Nc + 63) / 64, (Mr + GTM - 1) / GTM);
      gemm_bf16_kernel<2,true><<<grd, 256, 0, stream>>>(A0, B0, K0, A1, B1, K1,
          b0, b1, Cc, Cb, ldcb, Mr, Nc, act, accum);
    }
  };

  const bf16* src2bf = bigH0 + (size_t)MBD;
  gemm(src2bf, wq_t, 512, nullptr, nullptr, 0, nullptr, nullptr,
       Qf, nullptr, 0, T_DIM * B_DIM, 64, 0, 0);
  gemm(bigH0, wkv_t, 512, nullptr, nullptr, 0, nullptr, nullptr,
       KVf, nullptr, 0, J_DIM * B_DIM, 128, 0, 0);

  attn_kernel<<<B_DIM * H_DIM, 128, 0, stream>>>(Qf, KVf, Pbuf, u_, v_, AVf, AVbf, B_DIM);

  gemm(AVbf, wout_t, 64, nullptr, nullptr, 0, nullptr, nullptr,
       F0, bigH1, 512, T_DIM * B_DIM, 512, 0, 0);

  long long lNBD = (long long)NBD;
  int gN = (int)((NBD + 255) / 256);
  cvt_bf16_kernel<<<gN, 256, 0, stream>>>(aug, bigH0, lNBD);   // xbf

  // ---- gate 1 (x = aug, y = F0) ----
  gemm(bigH1, g1_t + 0*DD, 512, bigH0, g1_t + 1*DD, 512, g1_b + 0,    g1_b + 512,
       F1, nullptr, 0, T_DIM * B_DIM, 512, 0, 0);
  gemm(bigH1, g1_t + 2*DD, 512, bigH0, g1_t + 3*DD, 512, g1_b + 1024, g1_b + 1536,
       F2, nullptr, 0, T_DIM * B_DIM, 512, 0, 0);
  gemm(bigH1, g1_t + 4*DD, 512, nullptr, nullptr, 0,     g1_b + 2048, g1_b + 2560,
       F3, nullptr, 0, T_DIM * B_DIM, 512, 0, 0);
  gelem1_kernel<<<gN, 256, 0, stream>>>(F1, F2, aug, bigH0, lNBD);   // rxbf
  gemm(bigH0, g1_t + 5*DD, 512, nullptr, nullptr, 0, nullptr, nullptr,
       F3, nullptr, 0, T_DIM * B_DIM, 512, 0, 1);                    // F3 += rx@W5
  gelem2_kernel<<<gN, 256, 0, stream>>>(F2, aug, F3, F4, lNBD);      // src -> F4

  // ---- FF ----
  ln_kernel<<<T_DIM * B_DIM, 128, 0, stream>>>(F4, ln2_g, ln2_b, bigH1);
  gemm(bigH1, ff1_t, 512, nullptr, nullptr, 0, ff_b1, nullptr,
       nullptr, ff1bf, 64, T_DIM * B_DIM, 64, 1, 0);                 // relu
  gemm(ff1bf, ff2_t, 64, nullptr, nullptr, 0, ff_b2, nullptr,
       F0, bigH1, 512, T_DIM * B_DIM, 512, 0, 0);                    // y2 + ybf2
  cvt_bf16_kernel<<<gN, 256, 0, stream>>>(F4, bigH0, lNBD);          // xbf2

  // ---- gate 2 (x = F4, y = F0) ----
  gemm(bigH1, g2_t + 0*DD, 512, bigH0, g2_t + 1*DD, 512, g2_b + 0,    g2_b + 512,
       F1, nullptr, 0, T_DIM * B_DIM, 512, 0, 0);
  gemm(bigH1, g2_t + 2*DD, 512, bigH0, g2_t + 3*DD, 512, g2_b + 1024, g2_b + 1536,
       F2, nullptr, 0, T_DIM * B_DIM, 512, 0, 0);
  gemm(bigH1, g2_t + 4*DD, 512, nullptr, nullptr, 0,     g2_b + 2048, g2_b + 2560,
       F3, nullptr, 0, T_DIM * B_DIM, 512, 0, 0);
  gelem1_kernel<<<gN, 256, 0, stream>>>(F1, F2, F4, bigH0, lNBD);
  gemm(bigH0, g2_t + 5*DD, 512, nullptr, nullptr, 0, nullptr, nullptr,
       F3, nullptr, 0, T_DIM * B_DIM, 512, 0, 1);
  gelem2_kernel<<<gN, 256, 0, stream>>>(F2, F4, F3, F1, lNBD);       // out -> F1

  // ---- pool + critic head ----
  pool_kernel<<<(512 * 512 + 255) / 256, 256, 0, stream>>>(F1, ts, tsbf, T_DIM,
                                                           (size_t)B_DIM * D_DIM);
  gemm(tsbf, d1_t, 512, nullptr, nullptr, 0, d1_b, nullptr,
       nullptr, cath, 1088, 512, 1024, 1, 0);
  actcat_kernel<<<(512 * 64 + 255) / 256, 256, 0, stream>>>(action, cath);
  gemm(cath, d2_t, 1088, nullptr, nullptr, 0, d2_b, nullptr,
       nullptr, h2bf, 512, 512, 512, 1, 0);
  gemm(h2bf, d3_t, 512, nullptr, nullptr, 0, d3_b, nullptr,
       h3, nullptr, 0, 512, 300, 1, 0);
  head4_kernel<<<2, 256, 0, stream>>>(h3, d4_W, d4_b, (float*)d_out, 512, 300);
}
